// Decoder_68247030334424
// MI455X (gfx1250) — compile-verified, matrix operand order checked by reading, not run on hardware
//
#include <hip/hip_runtime.h>
#include <hip/hip_bf16.h>

// ---------------------------------------------------------------------------
// Tacotron decoder, persistent-kernel formulation for MI455X (gfx1250).
//  * 8 workgroups x 512 threads (16 wave32s); WG b owns batch rows [16b,16b+16)
//  * 200-step recurrence runs inside the kernel (batch rows are independent,
//    so no cross-WG sync is ever needed)
//  * all GEMMs via v_wmma_f32_16x16x32_f16 (f16 in, f32 accumulate)
//  * GRU gates fused in registers (6 accumulators / wave), activations VALU
//  * memory tensor (64MB) stays fp32: it is L2-resident (192MB L2)
//  * weight B-fragments are t-loop invariant; compiler register-caches what
//    fits and spills the rest once (reload ~= one VMEM per step) -- fine
//  * A-fragments double-buffered so ds_load latency overlaps the WMMAs
// ---------------------------------------------------------------------------

typedef _Float16 half8  __attribute__((ext_vector_type(8)));
typedef _Float16 v16h   __attribute__((ext_vector_type(16)));
typedef float    v8f    __attribute__((ext_vector_type(8)));

#define WMMA(a, b, c) \
  __builtin_amdgcn_wmma_f32_16x16x32_f16(false, (a), false, (b), (short)0, (c), false, false)

#if __has_builtin(__builtin_amdgcn_sched_barrier)
#define SCHED_FENCE() __builtin_amdgcn_sched_barrier(0)
#else
#define SCHED_FENCE()
#endif

// LDS leading dims (elements). f16 ones are multiples of 8, f32 ones of 4 (16B-aligned b128 frags)
// and row strides are ≡16B (mod 256B) -> 4-bank rotation, no 16-way conflicts.
#define LDX  520   // f16 staging A (x pad416 / [ctx|query] K=512)
#define LDY  264   // f16 staging A (prenet1 out / proj out)
#define LDM  264   // f16 state mirrors (query, h1, h2)
#define LDF  260   // f32 state / scratch
#define LDSC 516   // f32 scores/attention weights (Te=512)

#define SMEM_BYTES ((size_t)((3*16*LDF + 16*LDF + 16*LDSC + 256) * 4 + \
                             (16*LDX + 16*LDY + 3*16*LDM) * 2))

__device__ __forceinline__ float sigm_f(float x) { return 1.f / (1.f + __expf(-x)); }
__device__ __forceinline__ float tanh_f(float x) {
#if __has_builtin(__builtin_amdgcn_tanhf)
  return __builtin_amdgcn_tanhf(x);      // v_tanh_f32 (gfx1250 TRANS op)
#else
  float e = __expf(2.f * x);
  return 1.f - 2.f / (e + 1.f);
#endif
}

// 16x32 f16 fragment: lane L -> row (L&15), halves (L>>4): K = {h*8..h*8+7, 16+h*8..}
__device__ __forceinline__ v16h load_frag(const _Float16* p0) {
  const half8* p = (const half8*)p0;
  half8 lo = p[0];
  half8 hi = p[2];   // +16 f16 elements
  return __builtin_shufflevector(lo, hi, 0,1,2,3,4,5,6,7,8,9,10,11,12,13,14,15);
}
__device__ __forceinline__ v16h loadA(const _Float16* base, int ld, int k0, int lane) {
  int m = lane & 15, h = (lane >> 4) * 8;
  return load_frag(base + m * ld + k0 + h);
}
// B (K x 16) fragment from row-major weight W[N][K]: lane n streams W[n][k..]
__device__ __forceinline__ v16h loadBw(const _Float16* __restrict__ W, int K,
                                       int nBase, int k0, int lane) {
  int n = nBase + (lane & 15), h = (lane >> 4) * 8;
  return load_frag(W + (size_t)n * K + k0 + h);
}

// A-fragment double-buffered: ds_load for k+32 is in flight while WMMA(k) runs.
__device__ __forceinline__ v8f gemm_tile(const _Float16* A, int ldA,
                                         const _Float16* __restrict__ W, int K,
                                         int nBase, int lane) {
  v8f acc = {};
  SCHED_FENCE();
  v16h a = loadA(A, ldA, 0, lane);
#pragma unroll 2
  for (int k = 0; k < K; k += 32) {
    v16h an = a;
    if (k + 32 < K) an = loadA(A, ldA, k + 32, lane);
    acc = WMMA(a, loadBw(W, K, nBase, k, lane), acc);
    a = an;
  }
  return acc;
}

// Fused GRU cell: wave w owns gate columns [16w,16w+16). All six gate tiles
// accumulate in registers; gate math + residual in registers; writes f32 state,
// f16 mirror (next WMMA A operand), optional extra f16 copy.
__device__ __forceinline__ void gru_fused(
    const _Float16* __restrict__ Wih, int Kx, const _Float16* Ax, int ldAx,
    const _Float16* __restrict__ Whh, const _Float16* Ah, int ldAh,
    const float* __restrict__ bih, const float* __restrict__ bhh,
    float* hState, int ldS, _Float16* hMirror, int ldM,
    const float* resid, int ldR,
    _Float16* extra, int ldE, int extraColOff,
    int w, int lane) {
  const int c = w * 16;
  v8f gir = {}, giz = {}, gin = {}, ghr = {}, ghz = {}, ghn = {};
  SCHED_FENCE();
  {
    v16h a = loadA(Ax, ldAx, 0, lane);
#pragma unroll 1
    for (int k = 0; k < Kx; k += 32) {
      v16h an = a;
      if (k + 32 < Kx) an = loadA(Ax, ldAx, k + 32, lane);
      gir = WMMA(a, loadBw(Wih, Kx, c,       k, lane), gir);
      giz = WMMA(a, loadBw(Wih, Kx, c + 256, k, lane), giz);
      gin = WMMA(a, loadBw(Wih, Kx, c + 512, k, lane), gin);
      a = an;
    }
  }
  {
    v16h a = loadA(Ah, ldAh, 0, lane);
#pragma unroll 1
    for (int k = 0; k < 256; k += 32) {
      v16h an = a;
      if (k + 32 < 256) an = loadA(Ah, ldAh, k + 32, lane);
      ghr = WMMA(a, loadBw(Whh, 256, c,       k, lane), ghr);
      ghz = WMMA(a, loadBw(Whh, 256, c + 256, k, lane), ghz);
      ghn = WMMA(a, loadBw(Whh, 256, c + 512, k, lane), ghn);
      a = an;
    }
  }
  const int n  = c + (lane & 15);
  const int mB = (lane >> 4) * 8;
  const float bir = bih[n], biz = bih[n + 256], bin_ = bih[n + 512];
  const float bhr = bhh[n], bhz = bhh[n + 256], bhn_ = bhh[n + 512];
  float hv[8];
#pragma unroll
  for (int r = 0; r < 8; ++r) {
    int m = mB + r;
    float rg = sigm_f(gir[r] + bir + ghr[r] + bhr);
    float zg = sigm_f(giz[r] + biz + ghz[r] + bhz);
    float ng = tanh_f(gin[r] + bin_ + rg * (ghn[r] + bhn_));
    float hnew = (1.f - zg) * ng + zg * hState[m * ldS + n];
    if (resid) hnew += resid[m * ldR + n];
    hv[r] = hnew;
  }
  __syncthreads();  // all waves done reading old mirror before overwrite
#pragma unroll
  for (int r = 0; r < 8; ++r) {
    int m = mB + r;
    hState[m * ldS + n]  = hv[r];
    hMirror[m * ldM + n] = (_Float16)hv[r];
    if (extra) extra[m * ldE + extraColOff + n] = (_Float16)hv[r];
  }
}

// --------------------------- prologue: f32 -> f16 (+pad) --------------------
__global__ void convpad_kernel(const float* __restrict__ src, _Float16* __restrict__ dst,
                               int Nsrc, int Ksrc, int Kdst, int total) {
  int i = blockIdx.x * blockDim.x + threadIdx.x;
  if (i >= total) return;
  int n = i / Kdst, k = i - n * Kdst;
  float v = (n < Nsrc && k < Ksrc) ? src[(size_t)n * Ksrc + k] : 0.f;
  dst[i] = (_Float16)v;
}

// --------------------------------- main -------------------------------------
__global__ void __launch_bounds__(512, 1)
decoder_kernel(const float* __restrict__ memory,   // (128,512,256)
               const float* __restrict__ target,   // (128,1000,80) = (128,200,400)
               const float* __restrict__ pb1, const float* __restrict__ pb2,
               const float* __restrict__ abih, const float* __restrict__ abhh,
               const float* __restrict__ vW,
               const float* __restrict__ g1bih, const float* __restrict__ g1bhh,
               const float* __restrict__ g2bih, const float* __restrict__ g2bhh,
               const _Float16* __restrict__ pW1h,   // 256x416
               const _Float16* __restrict__ pW2h,   // 128x256
               const _Float16* __restrict__ aWihh,  // 768x128
               const _Float16* __restrict__ aWhhh,  // 768x256
               const _Float16* __restrict__ qWh,    // 256x256
               const _Float16* __restrict__ projWh, // 256x512
               const _Float16* __restrict__ g1Wihh, const _Float16* __restrict__ g1Whhh,
               const _Float16* __restrict__ g2Wihh, const _Float16* __restrict__ g2Whhh,
               const _Float16* __restrict__ specWh, // 416x256
               float* __restrict__ out) {
  extern __shared__ char smem[];
  float*    stateF = (float*)smem;                    // 3 x 16 x LDF (query,h1,h2)
  float*    sOut   = stateF + 3 * 16 * LDF;           // 16 x LDF   (qp / proj-out)
  float*    sS     = sOut + 16 * LDF;                 // 16 x LDSC  (scores/attn)
  float*    sV     = sS + 16 * LDSC;                  // 256        (vW copy)
  _Float16* sX     = (_Float16*)(sV + 256);           // 16 x LDX
  _Float16* sY     = sX + 16 * LDX;                   // 16 x LDY
  _Float16* sHh    = sY + 16 * LDY;                   // 3 x 16 x LDM (f16 mirrors)

  const int tid   = threadIdx.x;
  const int w     = tid >> 5;
  const int lane  = tid & 31;
  const int bBase = blockIdx.x * 16;

  for (int i = tid; i < 3 * 16 * LDF; i += 512) stateF[i] = 0.f;
  for (int i = tid; i < 3 * 16 * LDM; i += 512) sHh[i] = (_Float16)0.f;
  for (int i = tid; i < 256; i += 512) sV[i] = vW[i];
  __syncthreads();

  const size_t TGT_SIZE = (size_t)128 * 200 * 400;

  for (int t = 0; t < 200; ++t) {
    // ---- 1. stage teacher-forcing input x (16x400 -> f16, pad K to 416) ----
    for (int i = tid; i < 16 * 416; i += 512) {
      int m = i / 416, k = i - m * 416;
      float v = 0.f;
      if (t > 0 && k < 400)
        v = target[(size_t)(bBase + m) * 80000 + (size_t)(t - 1) * 400 + k];
      sX[m * LDX + k] = (_Float16)v;
    }
    __syncthreads();

    // ---- 2. prenet1: relu(x @ pW1.T + pb1), N=256 (1 tile/wave) ----
    {
      const int c = w * 16;
      v8f acc = gemm_tile(sX, LDX, pW1h, 416, c, lane);
      const int n = c + (lane & 15), mB = (lane >> 4) * 8;
      const float b1 = pb1[n];
#pragma unroll
      for (int r = 0; r < 8; ++r)
        sY[(mB + r) * LDY + n] = (_Float16)fmaxf(acc[r] + b1, 0.f);
    }
    __syncthreads();

    // ---- 3. prenet2: relu(h1 @ pW2.T + pb2), N=128 (waves 0..7) ----
    if (w < 8) {
      const int c = w * 16;
      v8f acc = gemm_tile(sY, LDY, pW2h, 256, c, lane);
      const int n = c + (lane & 15), mB = (lane >> 4) * 8;
      const float b2 = pb2[n];
#pragma unroll
      for (int r = 0; r < 8; ++r)
        sX[(mB + r) * LDX + n] = (_Float16)fmaxf(acc[r] + b2, 0.f);
    }
    __syncthreads();

    // ---- 4. attention GRU: query = GRU(p, cell_h); f16 copy into sX[256..512) ----
    gru_fused(aWihh, 128, sX, LDX, aWhhh, sHh, LDM, abih, abhh,
              stateF, LDF, sHh, LDM, nullptr, 0, sX, LDX, 256, w, lane);
    __syncthreads();

    // ---- 5. qp = query @ qW.T (f32 into sOut) ----
    {
      const int c = w * 16;
      v8f acc = gemm_tile(sHh, LDM, qWh, 256, c, lane);
      const int n = c + (lane & 15), mB = (lane >> 4) * 8;
#pragma unroll
      for (int r = 0; r < 8; ++r) sOut[(mB + r) * LDF + n] = acc[r];
    }
    __syncthreads();

    // ---- 6. scores[b][te] = sum_h v[h]*tanh(qp[b][h] + mem[b][te][h]) ----
    SCHED_FENCE();
    for (int p = tid; p < 16 * 512; p += 512) {
      int b = p & 15, te = p >> 4;
      const float* mrow = memory + (((size_t)(bBase + b)) * 512 + te) * 256;
      const float* qrow = sOut + b * LDF;
      float s = 0.f;
#pragma unroll 4
      for (int h = 0; h < 256; h += 4) {
        float4 mv = *(const float4*)(mrow + h);
        s += sV[h + 0] * tanh_f(qrow[h + 0] + mv.x);
        s += sV[h + 1] * tanh_f(qrow[h + 1] + mv.y);
        s += sV[h + 2] * tanh_f(qrow[h + 2] + mv.z);
        s += sV[h + 3] * tanh_f(qrow[h + 3] + mv.w);
      }
      sS[b * LDSC + te] = s;
    }
    __syncthreads();

    // ---- 7. softmax (wave w <-> batch row w) + write attention output ----
    {
      const int b = w;
      float mx = -1e30f;
      for (int te = lane; te < 512; te += 32) mx = fmaxf(mx, sS[b * LDSC + te]);
      for (int off = 16; off; off >>= 1) mx = fmaxf(mx, __shfl_xor(mx, off, 32));
      float sum = 0.f;
      for (int te = lane; te < 512; te += 32) {
        float e = __expf(sS[b * LDSC + te] - mx);
        sS[b * LDSC + te] = e;
        sum += e;
      }
      for (int off = 16; off; off >>= 1) sum += __shfl_xor(sum, off, 32);
      const float inv = 1.f / sum;
      float* attnOut = out + TGT_SIZE + (((size_t)(bBase + b)) * 200 + t) * 512;
      for (int te = lane; te < 512; te += 32) {
        float a = sS[b * LDSC + te] * inv;
        sS[b * LDSC + te] = a;
        attnOut[te] = a;
      }
    }
    __syncthreads();

    // ---- 8. ctx[b] = sum_te aw[b][te]*mem[b][te][:]  (f16 -> sX[0..256)) ----
    SCHED_FENCE();
    {
      const int b = w;
      const int h0 = lane * 8;
      float a0 = 0, a1 = 0, a2 = 0, a3 = 0, a4 = 0, a5 = 0, a6 = 0, a7 = 0;
      const float* mrow0 = memory + ((size_t)(bBase + b)) * 512 * 256 + h0;
#pragma unroll 2
      for (int te = 0; te < 512; ++te) {
        float a = sS[b * LDSC + te];
        const float4* mp = (const float4*)(mrow0 + (size_t)te * 256);
        float4 m0 = mp[0], m1 = mp[1];
        a0 += a * m0.x; a1 += a * m0.y; a2 += a * m0.z; a3 += a * m0.w;
        a4 += a * m1.x; a5 += a * m1.y; a6 += a * m1.z; a7 += a * m1.w;
      }
      _Float16* dst = sX + b * LDX + h0;
      dst[0] = (_Float16)a0; dst[1] = (_Float16)a1; dst[2] = (_Float16)a2; dst[3] = (_Float16)a3;
      dst[4] = (_Float16)a4; dst[5] = (_Float16)a5; dst[6] = (_Float16)a6; dst[7] = (_Float16)a7;
    }
    __syncthreads();

    // ---- 9. out = [ctx|query] @ projW.T (K=512) ----
    {
      const int c = w * 16;
      v8f acc = gemm_tile(sX, LDX, projWh, 512, c, lane);
      const int n = c + (lane & 15), mB = (lane >> 4) * 8;
#pragma unroll
      for (int r = 0; r < 8; ++r) {
        sOut[(mB + r) * LDF + n] = acc[r];
        sY[(mB + r) * LDY + n]   = (_Float16)acc[r];
      }
    }
    __syncthreads();

    // ---- 10. h1 = GRU1(out, h1) + out ----
    gru_fused(g1Wihh, 256, sY, LDY, g1Whhh, sHh + 16 * LDM, LDM, g1bih, g1bhh,
              stateF + 16 * LDF, LDF, sHh + 16 * LDM, LDM,
              sOut, LDF, nullptr, 0, 0, w, lane);
    __syncthreads();

    // ---- 11. h2 = GRU2(h1, h2) + h1 ----
    gru_fused(g2Wihh, 256, sHh + 16 * LDM, LDM, g2Whhh, sHh + 32 * LDM, LDM,
              g2bih, g2bhh, stateF + 32 * LDF, LDF, sHh + 32 * LDM, LDM,
              stateF + 16 * LDF, LDF, nullptr, 0, 0, w, lane);
    __syncthreads();

    // ---- 12. tchar = h2 @ specW.T (N padded 400->416, 26 tiles) ----
    for (int tile = w; tile < 26; tile += 16) {
      const int c = tile * 16;
      v8f acc = gemm_tile(sHh + 32 * LDM, LDM, specWh, 256, c, lane);
      const int n = c + (lane & 15), mB = (lane >> 4) * 8;
      if (n < 400) {
#pragma unroll
        for (int r = 0; r < 8; ++r)
          out[(((size_t)(bBase + mB + r)) * 200 + t) * 400 + n] = acc[r];
      }
    }
    __syncthreads();
  }
}

// --------------------------------- launch -----------------------------------
extern "C" void kernel_launch(void* const* d_in, const int* in_sizes, int n_in,
                              void* d_out, int out_size, void* d_ws, size_t ws_size,
                              hipStream_t stream) {
  (void)in_sizes; (void)n_in; (void)out_size; (void)ws_size;
  const float* memory = (const float*)d_in[0];
  const float* target = (const float*)d_in[1];
  const float* pW1    = (const float*)d_in[2];
  const float* pb1    = (const float*)d_in[3];
  const float* pW2    = (const float*)d_in[4];
  const float* pb2    = (const float*)d_in[5];
  const float* aWih   = (const float*)d_in[6];
  const float* aWhh   = (const float*)d_in[7];
  const float* abih   = (const float*)d_in[8];
  const float* abhh   = (const float*)d_in[9];
  const float* qW     = (const float*)d_in[10];
  const float* vW     = (const float*)d_in[11];
  const float* projW  = (const float*)d_in[12];
  const float* g1Wih  = (const float*)d_in[13];
  const float* g1Whh  = (const float*)d_in[14];
  const float* g1bih  = (const float*)d_in[15];
  const float* g1bhh  = (const float*)d_in[16];
  const float* g2Wih  = (const float*)d_in[17];
  const float* g2Whh  = (const float*)d_in[18];
  const float* g2bih  = (const float*)d_in[19];
  const float* g2bhh  = (const float*)d_in[20];
  const float* specW  = (const float*)d_in[21];

  char* ws = (char*)d_ws;
  auto carve = [&ws](size_t elems) {
    _Float16* p = (_Float16*)ws;
    ws += elems * sizeof(_Float16);
    return p;
  };
  _Float16* pW1h   = carve(256 * 416);
  _Float16* pW2h   = carve(128 * 256);
  _Float16* aWihh  = carve(768 * 128);
  _Float16* aWhhh  = carve(768 * 256);
  _Float16* qWh    = carve(256 * 256);
  _Float16* projWh = carve(256 * 512);
  _Float16* g1Wihh = carve(768 * 256);
  _Float16* g1Whhh = carve(768 * 256);
  _Float16* g2Wihh = carve(768 * 256);
  _Float16* g2Whhh = carve(768 * 256);
  _Float16* specWh = carve(416 * 256);

  auto conv = [&](const float* s, _Float16* d, int Ns, int Ks, int Nd, int Kd) {
    int tot = Nd * Kd;
    convpad_kernel<<<(tot + 255) / 256, 256, 0, stream>>>(s, d, Ns, Ks, Kd, tot);
  };
  conv(pW1,   pW1h,   256, 400, 256, 416);
  conv(pW2,   pW2h,   128, 256, 128, 256);
  conv(aWih,  aWihh,  768, 128, 768, 128);
  conv(aWhh,  aWhhh,  768, 256, 768, 256);
  conv(qW,    qWh,    256, 256, 256, 256);
  conv(projW, projWh, 256, 512, 256, 512);
  conv(g1Wih, g1Wihh, 768, 256, 768, 256);
  conv(g1Whh, g1Whhh, 768, 256, 768, 256);
  conv(g2Wih, g2Wihh, 768, 256, 768, 256);
  conv(g2Whh, g2Whhh, 768, 256, 768, 256);
  conv(specW, specWh, 400, 256, 416, 256);

  hipFuncSetAttribute((const void*)decoder_kernel,
                      hipFuncAttributeMaxDynamicSharedMemorySize, (int)SMEM_BYTES);
  decoder_kernel<<<dim3(8), dim3(512), SMEM_BYTES, stream>>>(
      memory, target, pb1, pb2, abih, abhh, vW, g1bih, g1bhh, g2bih, g2bhh,
      pW1h, pW2h, aWihh, aWhhh, qWh, projWh, g1Wihh, g1Whhh, g2Wihh, g2Whhh,
      specWh, (float*)d_out);
}
